// Model_34316788695621
// MI455X (gfx1250) — matrix operand order repaired and checked by reading, not compiled, not run on hardware
//
#include <hip/hip_runtime.h>

// ---------------- problem constants (from reference) ----------------
#define NATOMS   20000
#define NFEAT    2784
#define PPSE     4        // n_pseudo
#define HID      256
#define NSTRUCT  500
// scale = (1/sqrt(P)) / AVG_ATOMS = 0.5 / 40
#define OUT_SCALE 0.0125f

// tiling
#define ROWS_PER_BLOCK 32                    // 2 row-tiles of 16
#define KSTEP 32                             // bf16 WMMA K
#define NBLOCKS (NATOMS / ROWS_PER_BLOCK)    // 625
#define NSTEP1 (NFEAT / KSTEP)               // 87
#define NSTEP2 (HID / KSTEP)                 // 8
// per (p, kstep) swizzled-weight chunk: 16 coltiles * 32 lanes * 16 bf16
#define CHUNK_ELEMS (16 * 32 * 16)           // 8192 bf16 = 16 KB
#define CHUNK_B128  (CHUNK_ELEMS / 8)        // 1024 x 16B

typedef __attribute__((ext_vector_type(16))) __bf16 v16bf;
typedef __attribute__((ext_vector_type(8)))  float  v8f;

// ---------------- bf16 helpers ----------------
__device__ __forceinline__ float bf2f(__bf16 x) {
    unsigned short u = __builtin_bit_cast(unsigned short, x);
    unsigned v = ((unsigned)u) << 16;
    return __builtin_bit_cast(float, v);
}
__device__ __forceinline__ __bf16 f2bf(float f) {
    unsigned u = __builtin_bit_cast(unsigned, f);
    u += 0x7FFFu + ((u >> 16) & 1u);          // round-to-nearest-even
    unsigned short s = (unsigned short)(u >> 16);
    return __builtin_bit_cast(__bf16, s);
}

// A fragment: 16x32 (MxK) row-major at base, row stride rs -> 2x ds_load_b128/lane.
__device__ __forceinline__ v16bf load_a_frag(const __bf16* A, int rs, int lane) {
    const int r  = lane & 15;
    const int kh = lane >> 4;
    const __bf16* Ar = A + r * rs;
    v16bf a;
#pragma unroll
    for (int i = 0; i < 8; ++i) a[i] = Ar[kh * 16 + i];
#pragma unroll
    for (int i = 0; i < 8; ++i) a[8 + i] = Ar[kh * 16 + 8 + i];
    return a;
}

__device__ __forceinline__ float silu(float x) {
    return x * (1.0f / (1.0f + __expf(-x)));
}

// ---------------- async global->LDS copy (CDNA5, ASYNCcnt) ----------------
__device__ __forceinline__ void async_b128(unsigned lds_addr, unsigned long long gaddr) {
    asm volatile("global_load_async_to_lds_b128 %0, %1, off"
                 :: "v"(lds_addr), "v"(gaddr) : "memory");
}
__device__ __forceinline__ void wait_async0()  { asm volatile("s_wait_asynccnt 0"  ::: "memory"); }
__device__ __forceinline__ void wait_async16() { asm volatile("s_wait_asynccnt 16" ::: "memory"); }

// issue one K-step's swizzled B slab (4 x 16KB chunks) as 16 async b128/thread
__device__ __forceinline__ void issueB(const __bf16* W, int s, int nsteps,
                                       unsigned ldsBase, int tid) {
#pragma unroll
    for (int j = 0; j < 16; ++j) {
        int idx   = j * 256 + tid;         // 0..4095 b128s
        int pp    = idx >> 10;
        int local = idx & 1023;
        unsigned long long g = (unsigned long long)(void*)W
            + ((unsigned long long)(pp * nsteps + s) * CHUNK_B128 + local) * 16ull;
        async_b128(ldsBase + (unsigned)idx * 16u, g);
    }
}

// ---------------- weight fp32 -> bf16 conversion + WMMA swizzle ----------------
// Layout per region: [p][kstep s][coltile ct][lane][16 bf16]
//   value = W[p][ s*32 + (lane>>4)*16 + i ][ ct*16 + (lane&15) ]
#define C1 (PPSE * NFEAT * HID)    // 2,850,816 (W1)
#define C2 (PPSE * HID * HID)      //   262,144 (W2 and W3)

__device__ __forceinline__ void swizzle_one(const float* __restrict__ W,
                                            __bf16* __restrict__ dst,
                                            int ro, int nsteps, int krows) {
    const int i    = ro & 15;
    const int lane = (ro >> 4) & 31;
    const int ct   = (ro >> 9) & 15;
    const int rest = ro >> 13;            // p*nsteps + s
    const int s    = rest % nsteps;
    const int p    = rest / nsteps;
    const int k    = s * KSTEP + (lane >> 4) * 16 + i;
    const int n    = ct * 16 + (lane & 15);
    dst[ro] = f2bf(W[((size_t)p * krows + k) * HID + n]);
}

__global__ void convert_weights_kernel(const float* __restrict__ W1,
                                       const float* __restrict__ W2,
                                       const float* __restrict__ W3,
                                       __bf16* __restrict__ wb) {
    const int total = C1 + 2 * C2;
    for (int o = blockIdx.x * blockDim.x + threadIdx.x; o < total;
         o += gridDim.x * blockDim.x) {
        if (o < C1)            swizzle_one(W1, wb,            o,            NSTEP1, NFEAT);
        else if (o < C1 + C2)  swizzle_one(W2, wb + C1,       o - C1,       NSTEP2, HID);
        else                   swizzle_one(W3, wb + C1 + C2,  o - C1 - C2,  NSTEP2, HID);
    }
}

__global__ void zero_out_kernel(float* out, int n) {
    int i = blockIdx.x * blockDim.x + threadIdx.x;
    if (i < n) out[i] = 0.0f;
}

// ---------------- fused alchemical-MLP kernel ----------------
// LDS: B0,B1 : 2 x [4][16][32][16] bf16 = 2 x 65536 B  (double-buffered weight slab)
//      Hlds  : [4][32][256] bf16       = 65536 B       (activations, in-place)
//      A0,A1 : 2 x [32][32] bf16       = 2 x 2048 B    (pipelined feature tile)
//      pw    : 32*4 f32 in 1024 B      (red scratch reuses A0)
#define B0_OFF 0
#define B1_OFF 65536
#define H_OFF  131072
#define A0_OFF (H_OFF + 65536)
#define A1_OFF (A0_OFF + 2048)
#define PW_OFF (A1_OFF + 2048)
#define LDS_BYTES (PW_OFF + 1024)

__global__ void __launch_bounds__(256)
alchemy_mlp_kernel(const float* __restrict__ features,
                   const float* __restrict__ W_comb,
                   const __bf16* __restrict__ W1b,
                   const __bf16* __restrict__ W2b,
                   const __bf16* __restrict__ W3b,
                   const float* __restrict__ W4,
                   const int* __restrict__ species,
                   const int* __restrict__ sidx,
                   float* __restrict__ out) {
    extern __shared__ __align__(16) char smem[];
    __bf16* B0   = (__bf16*)(smem + B0_OFF);
    __bf16* B1   = (__bf16*)(smem + B1_OFF);
    __bf16* Hlds = (__bf16*)(smem + H_OFF);
    __bf16* A0   = (__bf16*)(smem + A0_OFF);
    __bf16* A1   = (__bf16*)(smem + A1_OFF);
    float*  pw   = (float*)(smem + PW_OFF);
    const unsigned bBase0 = (unsigned)(unsigned long long)(void*)B0;
    const unsigned bBase1 = (unsigned)(unsigned long long)(void*)B1;

    const int tid     = threadIdx.x;
    const int lane    = tid & 31;
    const int wave    = tid >> 5;          // 0..7
    const int p       = wave >> 1;         // pseudo species owned by this wave
    const int rowBase = blockIdx.x * ROWS_PER_BLOCK;

    if (tid < ROWS_PER_BLOCK * PPSE) {     // pw[r*4+pp] = W_comb[species[row]][pp]
        int r = tid >> 2, pp = tid & 3;
        pw[tid] = W_comb[species[rowBase + r] * PPSE + pp];
    }

    v8f acc[2][8];
    const v8f zv = {0.f, 0.f, 0.f, 0.f, 0.f, 0.f, 0.f, 0.f};

    // per-thread A-staging coordinates
    const int ar  = tid >> 3;              // row 0..31
    const int ac4 = (tid & 7) * 4;         // k offset 0..28
    const float* featRow = features + (size_t)(rowBase + ar) * NFEAT + ac4;

    // =========================== layer 1 ===========================
#pragma unroll
    for (int rt = 0; rt < 2; ++rt)
#pragma unroll
        for (int t = 0; t < 8; ++t) acc[rt][t] = zv;

    issueB(W1b, 0, NSTEP1, bBase0, tid);             // prologue: slab 0
    float4 f4cur = *(const float4*)featRow;          // prologue: A tile 0
    float4 f4nxt = f4cur;

    for (int s = 0; s < NSTEP1; ++s) {
        const int cur = s & 1;
        const bool more = (s + 1 < NSTEP1);
        if (more) issueB(W1b, s + 1, NSTEP1, cur ? bBase0 : bBase1, tid);
        {   // stage current A tile (converted) into Abuf[cur]
            __bf16* dst = (cur ? A1 : A0) + ar * KSTEP + ac4;
            dst[0] = f2bf(f4cur.x); dst[1] = f2bf(f4cur.y);
            dst[2] = f2bf(f4cur.z); dst[3] = f2bf(f4cur.w);
        }
        if (more) f4nxt = *(const float4*)(featRow + (s + 1) * KSTEP);
        if (more) wait_async16(); else wait_async0();
        __syncthreads();

        const __bf16* Ap = cur ? A1 : A0;
        const __bf16* Bp = cur ? B1 : B0;
        v16bf a0 = load_a_frag(Ap,              KSTEP, lane);
        v16bf a1 = load_a_frag(Ap + 16 * KSTEP, KSTEP, lane);
#pragma unroll
        for (int t = 0; t < 8; ++t) {
            int ct = (wave & 1) * 8 + t;
            v16bf b = *(const v16bf*)(Bp + ((p * 16 + ct) * 32 + lane) * 16);
            acc[0][t] = __builtin_amdgcn_wmma_f32_16x16x32_bf16(
                false, a0, false, b, (short)0, acc[0][t], false, false);
            acc[1][t] = __builtin_amdgcn_wmma_f32_16x16x32_bf16(
                false, a1, false, b, (short)0, acc[1][t], false, false);
        }
        __syncthreads();
        f4cur = f4nxt;
    }
    // epilogue 1: scale by pseudo weight, SiLU, write bf16 activations
#pragma unroll
    for (int rt = 0; rt < 2; ++rt)
#pragma unroll
        for (int t = 0; t < 8; ++t) {
            int col0 = ((wave & 1) * 8 + t) * 16;
#pragma unroll
            for (int i = 0; i < 8; ++i) {
                int rl = rt * 16 + (lane >> 4) * 8 + i;
                float v = acc[rt][t][i] * pw[rl * 4 + p];
                v = silu(v);
                Hlds[(p * ROWS_PER_BLOCK + rl) * HID + col0 + (lane & 15)] = f2bf(v);
            }
        }
    __syncthreads();

    // =========================== layers 2 & 3 ===========================
    for (int L = 0; L < 2; ++L) {
        const __bf16* Wb = (L == 0) ? W2b : W3b;
#pragma unroll
        for (int rt = 0; rt < 2; ++rt)
#pragma unroll
            for (int t = 0; t < 8; ++t) acc[rt][t] = zv;

        issueB(Wb, 0, NSTEP2, bBase0, tid);
        for (int s = 0; s < NSTEP2; ++s) {
            const int cur = s & 1;
            const bool more = (s + 1 < NSTEP2);
            if (more) issueB(Wb, s + 1, NSTEP2, cur ? bBase0 : bBase1, tid);
            if (more) wait_async16(); else wait_async0();
            __syncthreads();

            const int k0 = s * KSTEP;
            const __bf16* Bp = cur ? B1 : B0;
            v16bf a0 = load_a_frag(Hlds + (p * ROWS_PER_BLOCK +  0) * HID + k0,
                                   HID, lane);
            v16bf a1 = load_a_frag(Hlds + (p * ROWS_PER_BLOCK + 16) * HID + k0,
                                   HID, lane);
#pragma unroll
            for (int t = 0; t < 8; ++t) {
                int ct = (wave & 1) * 8 + t;
                v16bf b = *(const v16bf*)(Bp + ((p * 16 + ct) * 32 + lane) * 16);
                acc[0][t] = __builtin_amdgcn_wmma_f32_16x16x32_bf16(
                    false, a0, false, b, (short)0, acc[0][t], false, false);
                acc[1][t] = __builtin_amdgcn_wmma_f32_16x16x32_bf16(
                    false, a1, false, b, (short)0, acc[1][t], false, false);
            }
            __syncthreads();
        }
        // epilogue: SiLU, overwrite activations in place (all reads done)
#pragma unroll
        for (int rt = 0; rt < 2; ++rt)
#pragma unroll
            for (int t = 0; t < 8; ++t) {
                int col0 = ((wave & 1) * 8 + t) * 16;
#pragma unroll
                for (int i = 0; i < 8; ++i) {
                    int rl = rt * 16 + (lane >> 4) * 8 + i;
                    float v = silu(acc[rt][t][i]);
                    Hlds[(p * ROWS_PER_BLOCK + rl) * HID + col0 + (lane & 15)]
                        = f2bf(v);
                }
            }
        __syncthreads();
    }

    // ========== layer 4 (GEMV, all 8 waves) + segment scatter-add ==========
    {
        const int row  = tid & 31;         // atom row
        const int seg  = wave;             // wave-uniform: (p, half) segment
        const int pp   = seg >> 1;
        const int half = seg & 1;
        const __bf16* h  = Hlds + (pp * ROWS_PER_BLOCK + row) * HID + half * 128;
        const float*  w4 = W4 + pp * HID + half * 128;   // wave-uniform -> s_load
        float partial = 0.0f;
        for (int j = 0; j < 128; ++j) partial += bf2f(h[j]) * w4[j];
        float* red = (float*)A0;           // 256 floats of scratch
        red[seg * 32 + row] = partial;
    }
    __syncthreads();
    if (tid < ROWS_PER_BLOCK) {
        const float* red = (const float*)A0;
        float s = 0.0f;
#pragma unroll
        for (int k = 0; k < 8; ++k) s += red[k * 32 + tid];
        atomicAdd(out + sidx[rowBase + tid], s * OUT_SCALE);
    }
}

// ---------------- host launcher ----------------
extern "C" void kernel_launch(void* const* d_in, const int* in_sizes, int n_in,
                              void* d_out, int out_size, void* d_ws, size_t ws_size,
                              hipStream_t stream) {
    const float* features = (const float*)d_in[0];
    const float* W_comb   = (const float*)d_in[1];
    const float* W1       = (const float*)d_in[2];
    const float* W2       = (const float*)d_in[3];
    const float* W3       = (const float*)d_in[4];
    const float* W4       = (const float*)d_in[5];
    const int*   species  = (const int*)d_in[6];
    const int*   sidx     = (const int*)d_in[7];
    float*       out      = (float*)d_out;

    __bf16* wb  = (__bf16*)d_ws;       // 6.75 MB of swizzled bf16 weights
    __bf16* W1b = wb;
    __bf16* W2b = wb + C1;
    __bf16* W3b = wb + C1 + C2;

    zero_out_kernel<<<1, 512, 0, stream>>>(out, NSTRUCT);
    convert_weights_kernel<<<2048, 256, 0, stream>>>(W1, W2, W3, wb);
    alchemy_mlp_kernel<<<NBLOCKS, 256, LDS_BYTES, stream>>>(
        features, W_comb, W1b, W2b, W3b, W4, species, sidx, out);
}
